// StyleConv2d_90314572300466
// MI455X (gfx1250) — matrix operand order repaired and checked by reading, not compiled, not run on hardware
//
#include <hip/hip_runtime.h>

// ---------------------------------------------------------------------------
// StyleGAN2 modulated conv for MI455X (gfx1250), bf16 WMMA implicit GEMM.
//
// B=16, CIN=COUT=512, H=W=64, 3x3, pad=1. conv ~309 GFLOP.
// bf16 WMMA peak ~2.5 PF/s -> ~120us compute; HBM floor ~280MB -> ~12us.
// Operand stream at 64x32 register tiles needs ~100+ TB/s, which only LDS
// (320KB/WGP) can supply -> LDS-staged GEMM, double buffered, with
// GLOBAL_LOAD_ASYNC_TO_LDS (ASYNCcnt) overlapping staging with the WMMAs.
//
//   style[b,i] = w[b,:].affine_w[i,:] + affine_b[i]
//   demod[b,o] = rsqrt( sum_i (sum_kh,kw W^2)[o,i] * style[b,i]^2 + eps )
//   Wpk (bf16) = W pre-swizzled into WMMA A-fragment order (ISA 7.12.2)
//   Xpk (bf16) = (input*style) transposed NHWC via LDS tile transpose
//   out        = [ implicit-GEMM conv(Wpk, Xpk) ] * demod + bias
//
// conv kernel: 8 waves, block tile 256(M) x 64(N = one row), wave tile 64x32,
// 8x v_wmma_f32_16x16x32_bf16 per K-chunk, 144 K-chunks (9 taps x 512/32).
// ---------------------------------------------------------------------------

#define NB    16
#define NCIN  512
#define NCOUT 512
#define NH    64
#define NSP   (NH * NH)          // 4096 spatial positions
#define NKC   144                // 9 taps * (512/32) K-chunks of 32
#define EPSD  1e-8f

#define ALSTR 24                 // A LDS lane stride (ushorts): 48B, bank-safe
#define BLSTR 40                 // B LDS position stride (ushorts): 80B, bank-safe
#define BPOS  68                 // 66 halo positions, padded

typedef __attribute__((ext_vector_type(16))) __bf16 v16bf;
typedef __attribute__((ext_vector_type(8)))  float  v8f;
typedef __attribute__((ext_vector_type(4)))  int    v4i;

// address-space-qualified vector types for the async-copy builtin
typedef __attribute__((address_space(1))) v4i gv4i;   // global
typedef __attribute__((address_space(3))) v4i lv4i;   // LDS

struct Frag32 { uint4 lo, hi; };   // 32 bytes == one 16xbf16 WMMA operand

#if __has_builtin(__builtin_amdgcn_global_load_async_to_lds_b128)
#define HAVE_ASYNC_LDS 1
#else
#define HAVE_ASYNC_LDS 0
#endif

// 16B global -> LDS copy: async DMA on gfx1250 toolchains that expose it,
// otherwise classic load+ds_store.
__device__ __forceinline__ void async_copy16(void* l, const void* g) {
#if HAVE_ASYNC_LDS
    __builtin_amdgcn_global_load_async_to_lds_b128(
        (gv4i*)g, (lv4i*)l, /*offset=*/0, /*cpol=*/0);
#else
    *(uint4*)l = *(const uint4*)g;
#endif
}

__device__ __forceinline__ void wait_async_copies() {
#if HAVE_ASYNC_LDS
    asm volatile("s_wait_asynccnt 0" ::: "memory");   // ISA 9.6 SOPP op 74
#endif
}

__device__ __forceinline__ unsigned short f2bf(float f) {
    // round-to-nearest-even fp32 -> bf16
    unsigned u = __builtin_bit_cast(unsigned, f);
    u += 0x7FFFu + ((u >> 16) & 1u);
    return (unsigned short)(u >> 16);
}

// ---------------- prologue kernels (cheap: <=4 MMAC each) -------------------

__global__ __launch_bounds__(256)
void style_kernel(const float* __restrict__ w, const float* __restrict__ aw,
                  const float* __restrict__ ab, float* __restrict__ style) {
    int idx = blockIdx.x * 256 + threadIdx.x;        // NB*NCIN = 8192
    int b = idx >> 9, i = idx & 511;
    float s = ab[i];
    const float* wr = w + b * 512;
    const float* ar = aw + i * 512;
    for (int j = 0; j < 512; ++j) s += wr[j] * ar[j];
    style[idx] = s;
}

__global__ __launch_bounds__(256)
void wsq_kernel(const float* __restrict__ weight, float* __restrict__ wsq) {
    int idx = blockIdx.x * 256 + threadIdx.x;        // NCOUT*NCIN = 262144
    const float* p = weight + (size_t)idx * 9;
    float s = 0.f;
    for (int k = 0; k < 9; ++k) { float x = p[k]; s += x * x; }
    wsq[idx] = s;
}

__global__ __launch_bounds__(256)
void demod_kernel(const float* __restrict__ wsq, const float* __restrict__ style,
                  float* __restrict__ dcoef) {
    int idx = blockIdx.x * 256 + threadIdx.x;        // NB*NCOUT = 8192
    int b = idx >> 9, o = idx & 511;
    const float* wr = wsq + o * 512;
    const float* sr = style + b * 512;
    float s = 0.f;
    for (int i = 0; i < 512; ++i) { float st = sr[i]; s += wr[i] * st * st; }
    dcoef[idx] = rsqrtf(s + EPSD);
}

// Repack weight (OIHW fp32) -> bf16 in WMMA A-fragment order.
// Layout: Wpk[ mt(32) ][ kc(144) ][ lane(32) ][ e(16) ], 32B contiguous/lane.
// ISA 16-bit A 16x32 layout: lane<16  -> M=lane,    K = {0..7,16..23}
//                            lane>=16 -> M=lane-16, K = {8..15,24..31}
__global__ __launch_bounds__(256)
void pack_w_kernel(const float* __restrict__ weight, unsigned short* __restrict__ Wpk) {
    int idx = blockIdx.x * 256 + threadIdx.x;        // 512*4608 = 2359296
    int e    = idx & 15;
    int lane = (idx >> 4) & 31;
    int kc   = (idx >> 9) % NKC;
    int mt   = idx / (NKC * 512);
    int M    = mt * 16 + (lane & 15);
    int base8 = (lane >> 4) * 8;
    int K    = (e < 8) ? (base8 + e) : (16 + base8 + (e - 8));
    int kpos = kc >> 4;                 // kh*3+kw
    int cblk = kc & 15;
    int ci   = cblk * 32 + K;
    Wpk[idx] = f2bf(weight[((size_t)M * 512 + ci) * 9 + kpos]);
}

// x_mod = input*style, NCHW fp32 -> NHWC bf16 via LDS 32x32 tile transpose.
// Reads: 128B coalesced along W; writes: 64B coalesced along C.
__global__ __launch_bounds__(256)
void pack_x_kernel(const float* __restrict__ input, const float* __restrict__ style,
                   unsigned short* __restrict__ Xpk) {
    __shared__ unsigned short tile[32][34];          // +2 pad: dodge bank conflicts
    const int b   = blockIdx.z;
    const int sp0 = blockIdx.x * 32;                 // spatial base (h*64+w)
    const int c0  = blockIdx.y * 32;                 // channel base
    const int tx  = threadIdx.x & 31;
    const int ty  = threadIdx.x >> 5;                // 8 rows per pass, 4 passes

#pragma unroll
    for (int i = 0; i < 32; i += 8) {
        const int ci = c0 + ty + i;
        const float v = input[((size_t)b * NCIN + ci) * NSP + sp0 + tx]
                        * style[b * NCIN + ci];
        tile[ty + i][tx] = f2bf(v);
    }
    __syncthreads();
#pragma unroll
    for (int i = 0; i < 32; i += 8) {
        const int sp = sp0 + ty + i;
        Xpk[((size_t)b * NSP + sp) * NCIN + c0 + tx] = tile[tx][ty + i];
    }
}

// ---------------- main LDS-staged implicit-GEMM conv kernel -----------------

__global__ __launch_bounds__(256)
void conv_wmma_kernel(const unsigned short* __restrict__ Wpk,
                      const unsigned short* __restrict__ Xpk,
                      const float* __restrict__ dcoef,
                      const float* __restrict__ bias,
                      float* __restrict__ out) {
    // double-buffered tiles: A = 256(M) x 32(K) in fragment order (padded),
    // B = one x-row slice [wx=-1..64][32 ch] with halo zeros materialized.
    __shared__ unsigned short smA[2][16 * 32 * ALSTR];   // 2 x 24 KB
    __shared__ unsigned short smB[2][BPOS * BLSTR];      // 2 x ~5.4 KB

    const int tid  = threadIdx.x;
    const int lane = tid & 31;
    const int wv   = tid >> 5;          // 8 waves
    const int wm   = wv & 3;            // 4 M-groups (64 rows each)
    const int wn   = wv >> 2;           // 2 N-groups (32 cols each)
    const int b    = blockIdx.z;
    const int h    = blockIdx.y;        // output row
    const int w0   = wn * 32;
    const int mtb0 = blockIdx.x * 16;   // block covers m-tiles mtb0..mtb0+15

    const int N     = lane & 15;
    const int khalf = lane >> 4;

    // ---- staging (async global -> LDS) -----------------------------------
    auto stageA = [&](int s, int kc) {
        const int mt = tid >> 4;        // 16 m-tiles, 16 threads each
        const int q  = tid & 15;
#pragma unroll
        for (int j = 0; j < 2; ++j) {   // 2 lanes per thread, 32B each
            const int ln = 2 * q + j;
            const unsigned short* g = Wpk +
                (((size_t)(mtb0 + mt) * NKC + kc) * 32 + ln) * 16;
            unsigned short* d = &smA[s][(mt * 32 + ln) * ALSTR];
            async_copy16(d, g);
            async_copy16(d + 8, g + 8);
        }
    };
    auto stageB = [&](int s, int kc) {
        const int kpos = kc >> 4, cblk = kc & 15;
        const int hy = h + kpos / 3 - 1;
        const bool rowok = (hy >= 0) && (hy < NH);
#pragma unroll
        for (int pass = 0; pass < 2; ++pass) {
            const int slot = tid + pass * 256;      // 272 slots of 16B
            if (slot < BPOS * 4) {
                const int p = slot >> 2, q = slot & 3;
                const int wx = p - 1;
                unsigned short* d = &smB[s][p * BLSTR + q * 8];
                if (rowok && wx >= 0 && wx < NH) {
                    async_copy16(d, Xpk +
                        ((size_t)(b * NSP + hy * NH + wx) * NCIN + cblk * 32 + q * 8));
                } else {
                    *(uint4*)d = uint4{0, 0, 0, 0}; // halo zeros via plain LDS store
                }
            }
        }
    };

    v8f acc[4][2] = {{v8f{}, v8f{}}, {v8f{}, v8f{}},
                     {v8f{}, v8f{}}, {v8f{}, v8f{}}};

    stageA(0, 0);
    stageB(0, 0);
    wait_async_copies();
    __syncthreads();

    for (int i = 0; i < NKC; ++i) {
        // issue async staging of next K-chunk; overlaps with WMMAs below
        if (i + 1 < NKC) {
            stageA((i + 1) & 1, i + 1);
            stageB((i + 1) & 1, i + 1);
        }
        const int s  = i & 1;
        const int kw = (i >> 4) % 3;

        // ---- 2 B fragments from LDS (halo zeros already in place) ----
        v16bf bmat[2];
#pragma unroll
        for (int n = 0; n < 2; ++n) {
            const int p = w0 + n * 16 + N + kw;     // wx+1 in [0,66)
            const uint4* bp = (const uint4*)&smB[s][p * BLSTR + khalf * 8];
            Frag32 f; f.lo = bp[0]; f.hi = bp[2];
            bmat[n] = __builtin_bit_cast(v16bf, f);
        }

        // ---- 4 A fragments x 2 B fragments -> 8 WMMA ----
#pragma unroll
        for (int t = 0; t < 4; ++t) {
            const uint4* ap = (const uint4*)&smA[s][((wm * 4 + t) * 32 + lane) * ALSTR];
            Frag32 f; f.lo = ap[0]; f.hi = ap[1];
            const v16bf amat = __builtin_bit_cast(v16bf, f);
#pragma unroll
            for (int n = 0; n < 2; ++n) {
                acc[t][n] = __builtin_amdgcn_wmma_f32_16x16x32_bf16(
                    /*neg_a=*/false, amat, /*neg_b=*/false, bmat[n],
                    /*c_mod=*/(short)0, acc[t][n],
                    /*reuse_a=*/false, /*reuse_b=*/false);
            }
        }

        wait_async_copies();            // my async stage for i+1 landed
        __syncthreads();                // whole block done staging & computing
    }

    // ---- epilogue: fuse demod + bias, store NCHW fp32 ----
    // D layout: VGPR r -> M = r (lanes 0-15) / M = 8+r (lanes 16-31), N=lane%16
    const int moff = khalf * 8;
#pragma unroll
    for (int t = 0; t < 4; ++t) {
        const int mt = mtb0 + wm * 4 + t;
#pragma unroll
        for (int r = 0; r < 8; ++r) {
            const int cout = mt * 16 + moff + r;
            const float dm = dcoef[b * NCOUT + cout];
            const float bs = bias[cout];
            float* orow = out + (((size_t)b * NCOUT + cout) * NH + h) * NH;
#pragma unroll
            for (int n = 0; n < 2; ++n) {
                orow[w0 + n * 16 + N] = acc[t][n][r] * dm + bs;
            }
        }
    }
}

// ---------------- host launcher ---------------------------------------------

extern "C" void kernel_launch(void* const* d_in, const int* in_sizes, int n_in,
                              void* d_out, int out_size, void* d_ws, size_t ws_size,
                              hipStream_t stream) {
    const float* input    = (const float*)d_in[0];  // [16,512,64,64]
    const float* w        = (const float*)d_in[1];  // [16,512]
    const float* affine_w = (const float*)d_in[2];  // [512,512]
    const float* affine_b = (const float*)d_in[3];  // [512]
    const float* weight   = (const float*)d_in[4];  // [512,512,3,3]
    const float* bias     = (const float*)d_in[5];  // [512]
    float* out            = (float*)d_out;          // [16,512,64,64]

    // workspace carve-up (~73 MB total)
    float* style = (float*)d_ws;                         // 8192 f32
    float* dcoef = style + NB * NCIN;                    // 8192 f32
    float* wsq   = dcoef + NB * NCOUT;                   // 262144 f32
    unsigned short* Wpk = (unsigned short*)(wsq + NCOUT * NCIN);   // 2359296 bf16
    unsigned short* Xpk = Wpk + (size_t)NCOUT * NCIN * 9;          // 33554432 bf16

    style_kernel<<<(NB * NCIN) / 256, 256, 0, stream>>>(w, affine_w, affine_b, style);
    wsq_kernel<<<(NCOUT * NCIN) / 256, 256, 0, stream>>>(weight, wsq);
    demod_kernel<<<(NB * NCOUT) / 256, 256, 0, stream>>>(wsq, style, dcoef);
    pack_w_kernel<<<((size_t)NCOUT * NCIN * 9) / 256, 256, 0, stream>>>(weight, Wpk);

    dim3 gx(NSP / 32, NCIN / 32, NB);  // (128, 16, 16) LDS tile transpose
    pack_x_kernel<<<gx, 256, 0, stream>>>(input, style, Xpk);

    dim3 grid(NCOUT / 256, NH, NB);    // (2, 64, 16)
    conv_wmma_kernel<<<grid, 256, 0, stream>>>(Wpk, Xpk, dcoef, bias, out);
}
